// CircularBottleneck_47639777247516
// MI455X (gfx1250) — matrix operand order repaired and checked by reading, not compiled, not run on hardware
//
#include <hip/hip_runtime.h>

// ---------------------------------------------------------------------------
// CircularBottleneck on MI455X (gfx1250, wave32, WMMA)
//
// conv(circular 17x17, masked) == cyclic conv on Z16xZ16 with a folded 16x16
// effective kernel.  Each conv layer is a GEMM:
//     out[n,oc,s] = sum_{u,ic} X[n,ic,s(+)u] * Weff[oc,ic,u]
// mapped onto v_wmma_f32_16x16x32_f16 (M=16 spatial rows, N=16 out-channels,
// K = 32 input channels per step, looped over 256 cyclic shifts u).
// ---------------------------------------------------------------------------

typedef _Float16 v8h  __attribute__((ext_vector_type(8)));
typedef _Float16 v16h __attribute__((ext_vector_type(16)));
typedef float    v8f  __attribute__((ext_vector_type(8)));

#define SPAT   256      // 16x16 spatial positions
#define BATCH  256
#define EPS    1e-5f
#define SLOPE  0.01f

// 27 (p, d) config pairs replicating config_list(16, 1):
// p = 1..7, d | 2p (ascending d), k_eff = 2p+1, support t in [8-p, 8+p] step d.
static __device__ const int CFG_P[27] = {1,1, 2,2,2, 3,3,3,3, 4,4,4,4,
                                         5,5,5,5, 6,6,6,6,6,6, 7,7,7,7};
static __device__ const int CFG_D[27] = {1,2, 1,2,4, 1,2,3,6, 1,2,4,8,
                                         1,2,5,10, 1,2,3,4,6,12, 1,2,7,14};

__device__ __forceinline__ bool mask1d(int t, int p, int d) {
    int lo = 8 - p;
    return (t >= lo) && (t <= 8 + p) && (((t - lo) % d) == 0);
}

__device__ __forceinline__ v16h cat16(v8h lo, v8h hi) {
    return __builtin_shufflevector(lo, hi, 0,1,2,3,4,5,6,7,8,9,10,11,12,13,14,15);
}

// ---------------------------------------------------------------------------
// Weight prep: fold 17x17 masked taps mod-16 into 16x16 cyclic kernel and
// emit f16 B-fragments in WMMA lane order.
// Fragment f = ((octile*256 + u)*KSTEPS + ks), 512 halves per fragment.
// Within a fragment: half index = lane*16 + q,  n = lane&15,
//                    k = (q>>3)*16 + (lane>>4)*8 + (q&7)   (B 32x16 f16 VGPR layout).
// ---------------------------------------------------------------------------
__device__ __forceinline__ void prep_one(long e, const float* __restrict__ w,
                                         _Float16* __restrict__ wf,
                                         int IC, int OC, int KSTEPS, int grp3) {
    int  r    = (int)(e & 511);
    long f    = e >> 9;
    int  lane = r >> 4, q = r & 15;
    int  n    = lane & 15;
    int  k    = ((q >> 3) << 4) + ((lane >> 4) << 3) + (q & 7);
    int  ks   = (int)(f % KSTEPS);
    int  u    = (int)((f / KSTEPS) & 255);
    int  t    = (int)(f / (KSTEPS * 256));
    int  oc   = t * 16 + n;
    int  ic   = ks * 32 + k;

    float val = 0.f;
    if (oc < OC && ic < IC) {
        int c = grp3 ? (oc / 3) : oc;       // conv3: 3 out-channels per config
        int p = CFG_P[c], d = CFG_D[c];
        int a = u >> 4, b = u & 15;
        // taps kh with (kh-8) mod 16 == a, kh in [0,16]  (<=2 candidates)
        int khs[2]; int nkh = 0;
        if (a + 8 <= 16) khs[nkh++] = a + 8;
        if (a - 8 >= 0)  khs[nkh++] = a - 8;
        int kws[2]; int nkw = 0;
        if (b + 8 <= 16) kws[nkw++] = b + 8;
        if (b - 8 >= 0)  kws[nkw++] = b - 8;
        for (int i = 0; i < nkh; ++i)
            for (int j = 0; j < nkw; ++j)
                if (mask1d(khs[i], p, d) && mask1d(kws[j], p, d))
                    val += w[(((long)oc * IC + ic) * 17 + khs[i]) * 17 + kws[j]];
    }
    wf[e] = (_Float16)val;
}

__global__ void prep_weights(const float* __restrict__ w1,
                             const float* __restrict__ w2,
                             const float* __restrict__ w3,
                             _Float16* __restrict__ wf1,
                             _Float16* __restrict__ wf2,
                             _Float16* __restrict__ wf3) {
    long gid = (long)blockIdx.x * 256 + threadIdx.x;
    const long N1 = 1536L << 9;   // 2 octiles * 256 u * 3 ksteps
    const long N2 =  512L << 9;   // 2 octiles * 256 u * 1 kstep
    const long N3 = 1536L << 9;   // 6 octiles * 256 u * 1 kstep
    if (gid < N1)                prep_one(gid,            w1, wf1, 81, 27, 3, 0);
    else if (gid < N1 + N2)      prep_one(gid - N1,       w2, wf2, 27, 27, 1, 0);
    else if (gid < N1 + N2 + N3) prep_one(gid - N1 - N2,  w3, wf3, 27, 81, 1, 1);
}

// ---------------------------------------------------------------------------
// Conv layer: one workgroup (16 wave32) per image.  Wave w owns spatial rows
// oh = w (M-tile of 16).  Accumulate over 256 shifts u x KSTEPS ic-chunks.
// LDS activation layout: [spatial][ICP] f16, row stride ROW = ICP+8 halves
// (odd multiple of 16B -> conflict-free b128 reads across 16 lanes).
// Deterministic BN partials per block (shfl + LDS tree, no float atomics).
// ---------------------------------------------------------------------------
template <int IC, int OC, int OCT, int KSTEPS, bool BNIN>
__global__ __launch_bounds__(512)
void conv_kernel(const float* __restrict__ in, const float* __restrict__ sbin,
                 const _Float16* __restrict__ wf, float* __restrict__ yout,
                 float* __restrict__ part) {
    constexpr int ICP = KSTEPS * 32;
    constexpr int ROW = ICP + 8;
    __shared__ __align__(16) _Float16 xs[SPAT * ROW];
    __shared__ float swave[16][OCT * 16][2];

    const int n   = blockIdx.x;
    const int tid = threadIdx.x;

    // Stage this image into LDS as f16 [s][ic]; apply previous BN + lReLU.
    for (int idx = tid; idx < SPAT * ICP; idx += 512) {
        int s = idx / ICP, ic = idx - s * ICP;
        float v = 0.f;
        if (ic < IC) {
            v = in[((long)n * IC + ic) * SPAT + s];
            if (BNIN) {
                v = v * sbin[ic] + sbin[96 + ic];
                v = v > 0.f ? v : SLOPE * v;
            }
        }
        xs[s * ROW + ic] = (_Float16)v;
    }
    __syncthreads();

    const int w    = tid >> 5;       // wave id = oh of this M-tile
    const int lane = tid & 31;
    const int m    = lane & 15;      // A-row / D-column index
    const int hi   = lane >> 4;

    v8f acc[OCT] = {};

    for (int u = 0; u < 256; ++u) {
        const int a  = u >> 4, b = u & 15;
        // shifted spatial position for this lane's A-row: s' = (oh+a, ow+b)
        const int s2 = (((w + a) & 15) << 4) | ((m + b) & 15);
        const _Float16* arow = xs + s2 * ROW + hi * 8;
#pragma unroll
        for (int ks = 0; ks < KSTEPS; ++ks) {
            // A 16x32 f16 layout: lanes0-15 K in {0..7,16..23}, lanes16-31 +8
            v8h alo = *(const v8h*)(arow + ks * 32);
            v8h ahi = *(const v8h*)(arow + ks * 32 + 16);
            v16h A  = cat16(alo, ahi);
#pragma unroll
            for (int t = 0; t < OCT; ++t) {
                const _Float16* bp =
                    wf + (((long)(t * 256 + u) * KSTEPS + ks) << 9) + lane * 16;
                v8h blo = *(const v8h*)(bp);
                v8h bhi = *(const v8h*)(bp + 8);
                v16h B  = cat16(blo, bhi);
                acc[t] = __builtin_amdgcn_wmma_f32_16x16x32_f16(
                    false, A, false, B, (short)0, acc[t], false, false);
            }
        }
    }

    // D 16x16 f32 layout: VGPR r -> M = r + hi*8, N = m.  Emit y + partials.
#pragma unroll
    for (int t = 0; t < OCT; ++t) {
        const int oc = t * 16 + m;
        float ps = 0.f, pq = 0.f;
#pragma unroll
        for (int r = 0; r < 8; ++r) {
            float v = acc[t][r];
            int   s = (w << 4) + (hi << 3) + r;
            if (oc < OC) yout[((long)n * OC + oc) * SPAT + s] = v;
            ps += v; pq += v * v;
        }
        ps += __shfl_xor(ps, 16, 32);
        pq += __shfl_xor(pq, 16, 32);
        if (hi == 0) { swave[w][oc][0] = ps; swave[w][oc][1] = pq; }
    }
    __syncthreads();
    if (tid < OCT * 16) {
        float s = 0.f, q = 0.f;
#pragma unroll 4
        for (int ww = 0; ww < 16; ++ww) { s += swave[ww][tid][0]; q += swave[ww][tid][1]; }
        part[(long)n * 192 + tid]      = s;
        part[(long)n * 192 + 96 + tid] = q;
    }
}

// Fold batch statistics into per-channel (scale, bias): BN(x) = x*sc + bi.
__global__ void stats_finalize(const float* __restrict__ part,
                               const float* __restrict__ g,
                               const float* __restrict__ b,
                               float* __restrict__ sb, int OC) {
    int c = threadIdx.x;
    if (c >= OC) return;
    float s = 0.f, q = 0.f;
    for (int n = 0; n < BATCH; ++n) {
        s += part[(long)n * 192 + c];
        q += part[(long)n * 192 + 96 + c];
    }
    const float inv  = 1.f / 65536.f;   // B*H*W
    float mean = s * inv;
    float var  = q * inv - mean * mean;
    float sc   = g[c] * rsqrtf(var + EPS);
    sb[c]      = sc;
    sb[96 + c] = b[c] - mean * sc;
}

// out = lrelu(BN3(y3) + x)
__global__ void final_kernel(const float* __restrict__ y3,
                             const float* __restrict__ sb3,
                             const float* __restrict__ x,
                             float* __restrict__ out, long N) {
    long i = (long)blockIdx.x * 256 + threadIdx.x;
    if (i >= N) return;
    int c   = (int)((i >> 8) % 81);
    float v = y3[i] * sb3[c] + sb3[96 + c] + x[i];
    out[i]  = v > 0.f ? v : SLOPE * v;
}

// ---------------------------------------------------------------------------
extern "C" void kernel_launch(void* const* d_in, const int* in_sizes, int n_in,
                              void* d_out, int out_size, void* d_ws, size_t ws_size,
                              hipStream_t stream) {
    (void)in_sizes; (void)n_in; (void)out_size; (void)ws_size;

    const float* x  = (const float*)d_in[0];
    const float* w1 = (const float*)d_in[1];
    const float* w2 = (const float*)d_in[2];
    const float* w3 = (const float*)d_in[3];
    const float* g1 = (const float*)d_in[4];
    const float* b1 = (const float*)d_in[5];
    const float* g2 = (const float*)d_in[6];
    const float* b2 = (const float*)d_in[7];
    const float* g3 = (const float*)d_in[8];
    const float* b3 = (const float*)d_in[9];
    float* out = (float*)d_out;

    char*  ws  = (char*)d_ws;
    size_t off = 0;
    auto alloc = [&](size_t bytes) -> void* {
        void* p = ws + off;
        off += (bytes + 255) & ~(size_t)255;
        return p;
    };
    _Float16* wf1 = (_Float16*)alloc((1536L << 9) * 2);   // 1.5 MB
    _Float16* wf2 = (_Float16*)alloc(( 512L << 9) * 2);   // 0.5 MB
    _Float16* wf3 = (_Float16*)alloc((1536L << 9) * 2);   // 1.5 MB
    float* y1 = (float*)alloc((long)BATCH * 27 * SPAT * 4);  // 6.75 MB
    float* y2 = (float*)alloc((long)BATCH * 27 * SPAT * 4);  // 6.75 MB
    float* y3 = (float*)alloc((long)BATCH * 81 * SPAT * 4);  // 20.25 MB
    float* p1 = (float*)alloc((long)BATCH * 192 * 4);
    float* p2 = (float*)alloc((long)BATCH * 192 * 4);
    float* p3 = (float*)alloc((long)BATCH * 192 * 4);
    float* sb1 = (float*)alloc(192 * 4);
    float* sb2 = (float*)alloc(192 * 4);
    float* sb3 = (float*)alloc(192 * 4);

    // Fold masks + 17x17->16x16 cyclic kernels, swizzle to WMMA B fragments.
    prep_weights<<<7168, 256, 0, stream>>>(w1, w2, w3, wf1, wf2, wf3);

    // conv1: 81ch -> 27ch (no input BN)
    conv_kernel<81, 27, 2, 3, false><<<BATCH, 512, 0, stream>>>(x,  nullptr, wf1, y1, p1);
    stats_finalize<<<1, 96, 0, stream>>>(p1, g1, b1, sb1, 27);
    // conv2: 27 -> 27, applies BN1+lrelu while staging input
    conv_kernel<27, 27, 2, 1, true ><<<BATCH, 512, 0, stream>>>(y1, sb1,    wf2, y2, p2);
    stats_finalize<<<1, 96, 0, stream>>>(p2, g2, b2, sb2, 27);
    // conv3: 27 -> 81, applies BN2+lrelu while staging input
    conv_kernel<27, 81, 6, 1, true ><<<BATCH, 512, 0, stream>>>(y2, sb2,    wf3, y3, p3);
    stats_finalize<<<1, 96, 0, stream>>>(p3, g3, b3, sb3, 81);

    const long N = (long)BATCH * 81 * SPAT;   // 5,308,416
    final_kernel<<<(int)((N + 255) / 256), 256, 0, stream>>>(y3, sb3, x, out, N);
}
